// GINNet_46084999086803
// MI455X (gfx1250) — matrix operand order repaired and checked by reading, not compiled, not run on hardware
//
#include <hip/hip_runtime.h>
#include <hip/hip_bf16.h>
#include <stdint.h>

// ---------------------------------------------------------------------------
// CDNA5 (gfx1250) GIN forward: bf16 WMMA matmuls, fused BN/ReLU epilogues,
// L2-resident atomic segment sums.
// ---------------------------------------------------------------------------

typedef __attribute__((ext_vector_type(16))) __bf16 v16bf;
typedef __attribute__((ext_vector_type(8)))  __bf16 v8bf;
typedef __attribute__((ext_vector_type(8)))  float  v8f;

#define HDIM 64
#define KMAX 128
#define MBLK 128          // rows per block (8 waves x 16-row strips)
#define LDSP (KMAX + 8)   // element pitch, keeps 16B alignment, breaks bank cycles

static __device__ __forceinline__ unsigned short f32_to_bf16_bits(float f) {
    union { float f; unsigned u; } x; x.f = f;
    unsigned r = x.u + 0x7FFFu + ((x.u >> 16) & 1u);   // round-to-nearest-even
    return (unsigned short)(r >> 16);
}

// ---------------------------------------------------------------------------
// Fused GEMM: Out[M,64] = epi( preA(A0 (+A1)) @ W[K,64] + bias )
//   preA: optional per-column affine (BN scale/shift) + optional ReLU
//   epi : optional ReLU
// Block: 256 thr = 8 waves; block tile 128x64; wave tile 16x64 (4 accumulators,
// A fragment reused across the 4 column tiles per K-step).
// ---------------------------------------------------------------------------
__global__ __launch_bounds__(256)
void gemm_bf16_wmma(const float* __restrict__ A0,
                    const float* __restrict__ A1,
                    const float* __restrict__ a_scale,
                    const float* __restrict__ a_shift,
                    int a_relu,
                    const float* __restrict__ W,     // [K][64] row-major f32
                    const float* __restrict__ bias,  // [64]
                    int out_relu,
                    float* __restrict__ Out,         // [M][64]
                    int M, int K)
{
    __shared__ alignas(16) unsigned short sA[MBLK * LDSP];   // [row][k] bf16 bits
    __shared__ alignas(16) unsigned short sB[HDIM * LDSP];   // [n][k]  bf16 bits (W^T)

    const int tid    = threadIdx.x;
    const int m_base = blockIdx.x * MBLK;

    // Stage W^T into LDS (K-contiguous per output column for b128 frag loads)
    for (int idx = tid; idx < HDIM * K; idx += 256) {
        int n = idx / K, k = idx - n * K;
        sB[n * LDSP + k] = f32_to_bf16_bits(W[k * HDIM + n]);
    }
    // Stage A tile with fused (h+agg) / BN-affine / ReLU, f32 -> bf16
    for (int idx = tid; idx < MBLK * K; idx += 256) {
        int r = idx / K, k = idx - r * K;
        int m = m_base + r;
        float v = 0.f;
        if (m < M) {
            v = A0[m * K + k];
            if (A1)      v += A1[m * K + k];
            if (a_scale) v = v * a_scale[k] + a_shift[k];
            if (a_relu)  v = fmaxf(v, 0.f);
        }
        sA[r * LDSP + k] = f32_to_bf16_bits(v);
    }
    __syncthreads();

    const int wave = tid >> 5;          // wave32
    const int lane = tid & 31;
    const int half = lane >> 4;
    const int l16  = lane & 15;
    const int rt   = wave * 16;         // wave row strip within block

    v8f c0 = {}, c1 = {}, c2 = {}, c3 = {};
    for (int k0 = 0; k0 < K; k0 += 32) {
        // A fragment (ISA 16-bit A layout): lane l16 = row;
        // lanes 0-15: K = {k0..k0+7, k0+16..k0+23}; lanes 16-31: other halves.
        const v8bf alo = *reinterpret_cast<const v8bf*>(&sA[(rt + l16) * LDSP + k0 + half * 8]);
        const v8bf ahi = *reinterpret_cast<const v8bf*>(&sA[(rt + l16) * LDSP + k0 + 16 + half * 8]);
        v16bf a;
        #pragma unroll
        for (int i = 0; i < 8; ++i) { a[i] = alo[i]; a[i + 8] = ahi[i]; }

        v8f* acc[4] = { &c0, &c1, &c2, &c3 };
        #pragma unroll
        for (int j = 0; j < 4; ++j) {
            // B fragment: lane l16 = col; K = {k0+half*8..+7, k0+16+half*8..+7} (same as A)
            const int n0 = j * 16;
            const v8bf blo = *reinterpret_cast<const v8bf*>(&sB[(n0 + l16) * LDSP + k0 + half * 8]);
            const v8bf bhi = *reinterpret_cast<const v8bf*>(&sB[(n0 + l16) * LDSP + k0 + 16 + half * 8]);
            v16bf b;
            #pragma unroll
            for (int i = 0; i < 8; ++i) { b[i] = blo[i]; b[i + 8] = bhi[i]; }
            // (neg_a, A, neg_b, B, c_mod, C, reuse_a, reuse_b)
            *acc[j] = __builtin_amdgcn_wmma_f32_16x16x32_bf16(false, a, false, b,
                                                              (short)0, *acc[j],
                                                              false, false);
        }
    }

    // Epilogue: bias + optional ReLU, D layout row_in_tile = half*8 + r
    const v8f* accs[4] = { &c0, &c1, &c2, &c3 };
    #pragma unroll
    for (int j = 0; j < 4; ++j) {
        const int colg = j * 16 + l16;
        const float bv = bias ? bias[colg] : 0.f;
        #pragma unroll
        for (int r = 0; r < 8; ++r) {
            int mg = m_base + rt + half * 8 + r;
            if (mg < M) {
                float v = (*accs[j])[r] + bv;
                if (out_relu) v = fmaxf(v, 0.f);
                Out[mg * HDIM + colg] = v;
            }
        }
    }
}

// ---------------------------------------------------------------------------
// Per-column training-mode BN stats -> fused affine: scale=g/sqrt(v+eps),
// shift=b-mean*scale. One block per feature column.
// ---------------------------------------------------------------------------
__global__ __launch_bounds__(256)
void col_stats(const float* __restrict__ X, int M, int F,
               const float* __restrict__ g, const float* __restrict__ b,
               float* __restrict__ scale, float* __restrict__ shift)
{
    __shared__ float s_sum[256], s_sq[256];
    const int f = blockIdx.x;
    float s = 0.f, q = 0.f;
    for (int m = threadIdx.x; m < M; m += 256) {
        float v = X[m * F + f];
        s += v; q += v * v;
    }
    s_sum[threadIdx.x] = s; s_sq[threadIdx.x] = q;
    __syncthreads();
    for (int off = 128; off > 0; off >>= 1) {
        if (threadIdx.x < off) {
            s_sum[threadIdx.x] += s_sum[threadIdx.x + off];
            s_sq[threadIdx.x]  += s_sq[threadIdx.x + off];
        }
        __syncthreads();
    }
    if (threadIdx.x == 0) {
        float inv = 1.f / (float)M;
        float mu  = s_sum[0] * inv;
        float var = s_sq[0] * inv - mu * mu;
        float is  = rsqrtf(var + 1e-5f);
        float sc  = g[f] * is;
        scale[f] = sc;
        shift[f] = b[f] - mu * sc;
    }
}

__global__ void fill_zero(float* __restrict__ p, int n) {
    int i = blockIdx.x * 256 + threadIdx.x;
    if (i < n) p[i] = 0.f;
}

// GIN aggregation: agg[dst] += h[src]  (one thread per (edge, feature))
__global__ __launch_bounds__(256)
void edge_scatter_add(const int* __restrict__ src, const int* __restrict__ dst,
                      const float* __restrict__ h, float* __restrict__ agg, int E)
{
    int t = blockIdx.x * 256 + threadIdx.x;
    int e = t >> 6, f = t & 63;
    if (e >= E) return;
    atomicAdd(&agg[dst[e] * HDIM + f], h[src[e] * HDIM + f]);
}

// global_add_pool: hg[batch[n]] += h[n]
__global__ __launch_bounds__(256)
void pool_add(const int* __restrict__ batch, const float* __restrict__ h,
              float* __restrict__ hg, int N)
{
    int t = blockIdx.x * 256 + threadIdx.x;
    int n = t >> 6, f = t & 63;
    if (n >= N) return;
    atomicAdd(&hg[batch[n] * HDIM + f], h[n * HDIM + f]);
}

// Head: out[g,c] = bn(hg2)[g,:] @ cls_w + cls_b   (C=10, too small for WMMA)
__global__ __launch_bounds__(256)
void head_cls(const float* __restrict__ hg2,
              const float* __restrict__ scale, const float* __restrict__ shift,
              const float* __restrict__ cw, const float* __restrict__ cb,
              float* __restrict__ out, int G)
{
    int t = blockIdx.x * 256 + threadIdx.x;
    if (t >= G * 10) return;
    int g = t / 10, c = t - g * 10;
    float acc = cb[c];
    #pragma unroll
    for (int k = 0; k < HDIM; ++k) {
        float v = hg2[g * HDIM + k] * scale[k] + shift[k];
        acc += v * cw[k * 10 + c];
    }
    out[t] = acc;
}

// ---------------------------------------------------------------------------
extern "C" void kernel_launch(void* const* d_in, const int* in_sizes, int n_in,
                              void* d_out, int out_size, void* d_ws, size_t ws_size,
                              hipStream_t stream)
{
    (void)n_in; (void)ws_size;
    const float* x          = (const float*)d_in[0];
    const int*   edge_index = (const int*)  d_in[1];
    const int*   batch      = (const int*)  d_in[2];
    const float* bn_feat_g  = (const float*)d_in[3];
    const float* bn_feat_b  = (const float*)d_in[4];
    const float* w_feat     = (const float*)d_in[5];
    const float* b_feat     = (const float*)d_in[6];
    const float* gin_w1     = (const float*)d_in[7];
    const float* gin_b1     = (const float*)d_in[8];
    const float* gin_bn_g   = (const float*)d_in[9];
    const float* gin_bn_b   = (const float*)d_in[10];
    const float* gin_w2     = (const float*)d_in[11];
    const float* gin_b2     = (const float*)d_in[12];
    const float* bn_fc_g    = (const float*)d_in[13];
    const float* bn_fc_b    = (const float*)d_in[14];
    const float* lin_w      = (const float*)d_in[15];
    const float* lin_b      = (const float*)d_in[16];
    const float* bn_h_g     = (const float*)d_in[17];
    const float* bn_h_b     = (const float*)d_in[18];
    const float* cls_w      = (const float*)d_in[19];
    const float* cls_b      = (const float*)d_in[20];
    float* out = (float*)d_out;

    const int N = in_sizes[2];          // nodes
    const int E = in_sizes[1] / 2;      // edges
    const int G = out_size / 10;        // graphs

    // workspace carve-up
    float* ws     = (float*)d_ws;
    float* h      = ws;                 ws += (size_t)N * HDIM;
    float* agg    = ws;                 ws += (size_t)N * HDIM;
    float* t      = ws;                 ws += (size_t)N * HDIM;
    float* hg     = ws;                 ws += (size_t)G * HDIM;
    float* hg2    = ws;                 ws += (size_t)G * HDIM;
    float* scaleA = ws;                 ws += 128;
    float* shiftA = ws;                 ws += 128;
    float* scaleT = ws;                 ws += 64;
    float* shiftT = ws;                 ws += 64;
    float* scaleC = ws;                 ws += 64;
    float* shiftC = ws;                 ws += 64;
    float* scaleD = ws;                 ws += 64;
    float* shiftD = ws;                 ws += 64;

    const int gemmBlocksN = (N + MBLK - 1) / MBLK;
    const int gemmBlocksG = (G + MBLK - 1) / MBLK;

    // 1) BN(x) folded into affine; 2) h = relu(bn(x) @ w_feat + b_feat)
    col_stats<<<128, 256, 0, stream>>>(x, N, 128, bn_feat_g, bn_feat_b, scaleA, shiftA);
    gemm_bf16_wmma<<<gemmBlocksN, 256, 0, stream>>>(
        x, nullptr, scaleA, shiftA, 0, w_feat, b_feat, 1, h, N, 128);

    // 3) GIN layers
    for (int i = 0; i < 3; ++i) {
        fill_zero<<<((N * HDIM) + 255) / 256, 256, 0, stream>>>(agg, N * HDIM);
        edge_scatter_add<<<((E * HDIM) + 255) / 256, 256, 0, stream>>>(
            edge_index, edge_index + E, h, agg, E);
        // t = (h + agg) @ W1 + b1
        gemm_bf16_wmma<<<gemmBlocksN, 256, 0, stream>>>(
            h, agg, nullptr, nullptr, 0,
            gin_w1 + i * HDIM * HDIM, gin_b1 + i * HDIM, 0, t, N, HDIM);
        // BN stats of t
        col_stats<<<HDIM, 256, 0, stream>>>(t, N, HDIM,
                                            gin_bn_g + i * HDIM, gin_bn_b + i * HDIM,
                                            scaleT, shiftT);
        // h = relu( relu(bn(t)) @ W2 + b2 )
        gemm_bf16_wmma<<<gemmBlocksN, 256, 0, stream>>>(
            t, nullptr, scaleT, shiftT, 1,
            gin_w2 + i * HDIM * HDIM, gin_b2 + i * HDIM, 1, h, N, HDIM);
    }

    // 4) global_add_pool
    fill_zero<<<((G * HDIM) + 255) / 256, 256, 0, stream>>>(hg, G * HDIM);
    pool_add<<<((N * HDIM) + 255) / 256, 256, 0, stream>>>(batch, h, hg, N);

    // 5) head: hg2 = relu(bn(hg) @ lin_w + lin_b); out = bn(hg2) @ cls_w + cls_b
    col_stats<<<HDIM, 256, 0, stream>>>(hg, G, HDIM, bn_fc_g, bn_fc_b, scaleC, shiftC);
    gemm_bf16_wmma<<<gemmBlocksG, 256, 0, stream>>>(
        hg, nullptr, scaleC, shiftC, 0, lin_w, lin_b, 1, hg2, G, HDIM);
    col_stats<<<HDIM, 256, 0, stream>>>(hg2, G, HDIM, bn_h_g, bn_h_b, scaleD, shiftD);
    head_cls<<<((G * 10) + 255) / 256, 256, 0, stream>>>(
        hg2, scaleD, shiftD, cls_w, cls_b, out, G);
}
